// CrossAttention_65446711657232
// MI455X (gfx1250) — matrix operand order repaired and checked by reading, not compiled, not run on hardware
//
#include <hip/hip_runtime.h>

// ---------------------------------------------------------------------------
// Types for CDNA5 WMMA (wave32): bf16 16x16x32, f32 accumulate
// ---------------------------------------------------------------------------
typedef __bf16 bf16_t;
typedef __attribute__((ext_vector_type(8)))  bf16_t   v8bf;
typedef __attribute__((ext_vector_type(16))) bf16_t   v16bf;
typedef __attribute__((ext_vector_type(8)))  float    v8f;
typedef __attribute__((ext_vector_type(4)))  uint32_t su4;
typedef __attribute__((ext_vector_type(8)))  uint32_t su8;

__device__ __forceinline__ bf16_t f2bf(float f) { return (bf16_t)f; }

// ---------------------------------------------------------------------------
// CDNA5 async global->LDS copy (ASYNCcnt path, ISA 15.18.3 op 98).
// VDST = 32-bit LDS byte address (generic LDS address truncates to addr[31:0],
// ISA 10.2 aperture rule), VADDR = 64-bit global address, GV mode.
// ---------------------------------------------------------------------------
__device__ __forceinline__ void async_g2l_b128(void* lds, const void* gaddr) {
  asm volatile("global_load_async_to_lds_b128 %0, %1, off"
               :
               : "v"((uint32_t)(uintptr_t)lds),
                 "v"((unsigned long long)(uintptr_t)gaddr)
               : "memory");
}
#define S_WAIT_ASYNC0() asm volatile("s_wait_asynccnt 0x0" ::: "memory")

// ---------------------------------------------------------------------------
// CDNA5 Tensor Data Mover: one 2D-tile DMA global->LDS per descriptor
// (ISA ch.7.2 / ch.8).  Groups 2/3 are NULL => 2D tensor semantics.
// D# group1 is loop-invariant; group0 carries the per-tile addresses.
//   - data_size = 1 (2-byte elements)
//   - pad_enable, pad_interval=3 (pad after every 64B), pad_amount=3 (4 dwords)
//     -> LDS row pitch 80B, exactly matching the [rows][40 x bf16] tiles.
// ---------------------------------------------------------------------------
__device__ __forceinline__ su8 tdm_group1(uint32_t dim0, uint32_t rowstride,
                                          uint32_t tile0, uint32_t tile1) {
  su8 g;
  g[0] = (1u << 16) | (1u << 20) | (3u << 22) | (3u << 25);  // size/pad ctrl
  g[1] = (dim0 & 0xFFFFu) << 16;                   // tensor_dim0 low16
  g[2] = (dim0 >> 16) | (0x8000u << 16);           // dim0 hi | tensor_dim1 lo
  g[3] = (tile0 & 0xFFFFu) << 16;                  // dim1 hi=0 | tile_dim0
  g[4] = tile1 & 0xFFFFu;                          // tile_dim1 | tile_dim2=0
  g[5] = rowstride;                                // tensor_dim0_stride low32
  g[6] = 0u;                                       // stride hi | dim1_stride lo
  g[7] = 0u;                                       // dim1_stride hi
  return g;
}

__device__ __forceinline__ void tdm_load_2d(void* lds, const void* gptr,
                                            su8 g1) {
  const unsigned long long ga = (unsigned long long)(uintptr_t)gptr;
  su4 g0;
  g0[0] = 1u;                                       // count=1 (valid), user
  g0[1] = (uint32_t)(uintptr_t)lds;                 // lds byte address
  g0[2] = (uint32_t)ga;                             // global_addr[31:0]
  g0[3] = ((uint32_t)(ga >> 32) & 0x01FFFFFFu) | (2u << 30);  // addr hi|type=2
  asm volatile("tensor_load_to_lds %0, %1" : : "s"(g0), "s"(g1) : "memory");
}

// A-matrix 16x32 bf16 fragment (ISA 7.12.2): lane = M row (lane&15),
// lanes 0-15 hold K = {0..7, 16..23}, lanes 16-31 hold K = {8..15, 24..31}.
__device__ __forceinline__ v16bf load_a_frag(const bf16_t* rowp, int sel) {
  v8bf lo = *reinterpret_cast<const v8bf*>(rowp + sel * 8);
  v8bf hi = *reinterpret_cast<const v8bf*>(rowp + 16 + sel * 8);
  return __builtin_shufflevector(lo, hi, 0, 1, 2, 3, 4, 5, 6, 7,
                                 8, 9, 10, 11, 12, 13, 14, 15);
}

// B-matrix 32x16 bf16 fragment: lane = N column (lane&15),
// lanes 0-15 hold K = {0..7, 16..23}, lanes 16-31 hold K = {8..15, 24..31}.
// colp points at a row of a [N][K] (pre-transposed) LDS tile.
__device__ __forceinline__ v16bf load_b_frag(const bf16_t* colp, int sel) {
  v8bf lo = *reinterpret_cast<const v8bf*>(colp + sel * 8);
  v8bf hi = *reinterpret_cast<const v8bf*>(colp + 16 + sel * 8);
  return __builtin_shufflevector(lo, hi, 0, 1, 2, 3, 4, 5, 6, 7,
                                 8, 9, 10, 11, 12, 13, 14, 15);
}

__device__ __forceinline__ v8f wmma_bf16(v16bf a, v16bf b, v8f c) {
  return __builtin_amdgcn_wmma_f32_16x16x32_bf16(false, a, false, b,
                                                 (short)0, c, false, false);
}

// ---------------------------------------------------------------------------
// f32 -> bf16 converts (plain and transposing for weights)
// ---------------------------------------------------------------------------
__global__ void cvt_f32_to_bf16(const float* __restrict__ in,
                                bf16_t* __restrict__ out, long long n) {
  long long i = (long long)blockIdx.x * blockDim.x + threadIdx.x;
  const long long st = (long long)gridDim.x * blockDim.x;
  for (; i < n; i += st) out[i] = f2bf(in[i]);
}

// W[K][N] f32 -> Wt[N][K] bf16 (so GEMM B-fragments are contiguous per lane)
__global__ void cvt_transpose_bf16(const float* __restrict__ W,
                                   bf16_t* __restrict__ Wt, int K, int N) {
  long long i = (long long)blockIdx.x * blockDim.x + threadIdx.x;
  const long long st = (long long)gridDim.x * blockDim.x;
  const long long total = (long long)K * N;
  for (; i < total; i += st) {
    const int k = (int)(i / N);
    const int n = (int)(i % N);
    Wt[(size_t)n * K + k] = f2bf(W[i]);
  }
}

// ---------------------------------------------------------------------------
// WMMA GEMM: C[M,N] = A[M,K](bf16 rowmajor) * Bt[N,K](bf16, B pre-transposed)
//            + bias[N].  256 threads = 8 waves; 128x128x32 tiles; each wave
//            computes 64(M) x 32(N) = 8 accumulators.  Double-buffered LDS
//            fed by the Tensor Data Mover (one descriptor per tile, issued by
//            wave 0, tracked on TENSORcnt) pipelined against WMMA.
// ---------------------------------------------------------------------------
constexpr int BM = 128, BN = 128, BK = 32, BKP = 40;  // BKP: LDS row pad

template <bool OUT_F32, bool TRANS_OUT>
__global__ __launch_bounds__(256) void gemm_bf16_wmma(
    const bf16_t* __restrict__ A, const bf16_t* __restrict__ Bt,
    const float* __restrict__ bias, void* __restrict__ Cout,
    int M, int N, int K) {
  __shared__ bf16_t As[2][BM][BKP];
  __shared__ bf16_t Bs[2][BN][BKP];

  const int tid = threadIdx.x;
  const int lane = tid & 31;
  const int wv = tid >> 5;
  const int lr16 = lane & 15;
  const int sel = lane >> 4;
  const int mb = blockIdx.y * BM;
  const int nb = blockIdx.x * BN;
  const int wm = (wv >> 2) * 64;  // wave M offset within block
  const int wn = (wv & 3) * 32;   // wave N offset within block

  const bf16_t* gA0 = A + (size_t)mb * K;   // A tile origin (k advances)
  const bf16_t* gB0 = Bt + (size_t)nb * K;  // B tile origin
  const su8 g1 = tdm_group1((uint32_t)K, (uint32_t)K, BK, BM);

  // Prologue: wave 0 issues the two tile-0 DMAs.
  if (wv == 0) {
    tdm_load_2d(&As[0][0][0], gA0, g1);
    tdm_load_2d(&Bs[0][0][0], gB0, g1);
  }

  v8f acc[4][2] = {};
  const int nk = K / BK;
  for (int t = 0; t < nk; ++t) {
    const int cur = t & 1;
    if (wv == 0) {
      if (t + 1 < nk) {
        // Issue next tile into the other buffer (safe: end-of-iter barrier of
        // the previous step guarantees nobody still reads it).
        tdm_load_2d(&As[cur ^ 1][0][0], gA0 + (size_t)(t + 1) * BK, g1);
        tdm_load_2d(&Bs[cur ^ 1][0][0], gB0 + (size_t)(t + 1) * BK, g1);
        // TDM ops retire in order per wave: <=2 outstanding => tile t resident
        __builtin_amdgcn_s_wait_tensorcnt((short)2);
      } else {
        __builtin_amdgcn_s_wait_tensorcnt((short)0);
      }
    }
    __syncthreads();  // tile t resident for every wave

    v16bf af[4], bfr[2];
#pragma unroll
    for (int mt = 0; mt < 4; ++mt)
      af[mt] = load_a_frag(&As[cur][wm + mt * 16 + lr16][0], sel);
#pragma unroll
    for (int nt = 0; nt < 2; ++nt)
      bfr[nt] = load_b_frag(&Bs[cur][wn + nt * 16 + lr16][0], sel);
#pragma unroll
    for (int mt = 0; mt < 4; ++mt)
#pragma unroll
      for (int nt = 0; nt < 2; ++nt)
        acc[mt][nt] = wmma_bf16(af[mt], bfr[nt], acc[mt][nt]);

    __syncthreads();  // everyone done reading tile t (its buffer is reused t+2)
  }

  // Epilogue. C/D layout: VGPR i -> row (i + 8*sel) of tile, col = lane&15.
#pragma unroll
  for (int nt = 0; nt < 2; ++nt) {
    const int cn = nb + wn + nt * 16 + lr16;
    const float bsv = bias ? bias[cn] : 0.0f;
#pragma unroll
    for (int mt = 0; mt < 4; ++mt) {
#pragma unroll
      for (int i = 0; i < 8; ++i) {
        const int rm = mb + wm + mt * 16 + i + 8 * sel;
        const float v = acc[mt][nt][i] + bsv;
        if (OUT_F32)
          reinterpret_cast<float*>(Cout)[(size_t)rm * N + cn] = v;
        else if (TRANS_OUT)
          reinterpret_cast<bf16_t*>(Cout)[(size_t)cn * M + rm] = f2bf(v);
        else
          reinterpret_cast<bf16_t*>(Cout)[(size_t)rm * N + cn] = f2bf(v);
      }
    }
  }
}

// ---------------------------------------------------------------------------
// Fused flash attention (bf16 WMMA, f32 online softmax in registers).
// Q:[B*SQ][E] bf16 (head h at cols h*64..), K:[B*SKV][E] bf16,
// Vt:[E][B*SKV] bf16 (transposed by V-GEMM), O:[B*SQ][E] bf16.
// Block = (qblock=128 rows, head, batch); 8 waves, each owns 16 q rows.
// Tile loads use the async global->LDS pipe (ASYNCcnt).
// ---------------------------------------------------------------------------
constexpr int HDIM = 64, BQ = 128, BKV = 64, HP = 72;  // HP: LDS row pad

__global__ __launch_bounds__(256) void attn_fwd_wmma(
    const bf16_t* __restrict__ Q, const bf16_t* __restrict__ Kg,
    const bf16_t* __restrict__ Vt, bf16_t* __restrict__ Og,
    int SQ, int SKV, int E, int KVTOT, float scale) {
  __shared__ bf16_t Qs[BQ][HP];
  __shared__ bf16_t Ks[BKV][HP];
  __shared__ bf16_t Vs[HDIM][HP];
  __shared__ bf16_t Ps[8][16][HP];  // wave-private P (C-layout -> A-layout)

  const int tid = threadIdx.x;
  const int lane = tid & 31;
  const int wv = tid >> 5;
  const int lr16 = lane & 15;
  const int sel = lane >> 4;
  const int qb = blockIdx.x, hh = blockIdx.y, bb = blockIdx.z;
  const size_t qrow0 = (size_t)bb * SQ + (size_t)qb * BQ;
  const size_t krow0 = (size_t)bb * SKV;
  const size_t hcol = (size_t)hh * HDIM;

  {  // async-load 128x64 Q tile
    const int r = tid >> 1, c = (tid & 1) * 32;
    const bf16_t* g = Q + (qrow0 + r) * E + hcol + c;
    bf16_t* l = &Qs[r][c];
    async_g2l_b128(l, g);
    async_g2l_b128(l + 8, g + 8);
    async_g2l_b128(l + 16, g + 16);
    async_g2l_b128(l + 24, g + 24);
  }
  S_WAIT_ASYNC0();
  __syncthreads();

  v16bf qa0, qa1;
  {
    const bf16_t* rp = &Qs[wv * 16 + lr16][0];
    qa0 = load_a_frag(rp, sel);       // d = 0..31
    qa1 = load_a_frag(rp + 32, sel);  // d = 32..63
  }

  v8f ov[4] = {};  // O accumulator: 16 rows x 64 d
  float m_run[8], l_run[8];
#pragma unroll
  for (int i = 0; i < 8; ++i) { m_run[i] = -INFINITY; l_run[i] = 0.0f; }

  const int kvr = tid >> 2;        // 0..63: K/V tile row this thread loads
  const int kvc = (tid & 3) * 16;  // 0/16/32/48
  const int nkv = SKV / BKV;
  for (int kc = 0; kc < nkv; ++kc) {
    __syncthreads();  // previous iteration done reading Ks/Vs
    {  // async-load K chunk [64 kv][64 d] and V chunk transposed [64 d][64 kv]
      const bf16_t* gk = Kg + (krow0 + (size_t)kc * BKV + kvr) * E + hcol + kvc;
      bf16_t* lk = &Ks[kvr][kvc];
      async_g2l_b128(lk, gk);
      async_g2l_b128(lk + 8, gk + 8);
      const bf16_t* gv =
          Vt + (hcol + kvr) * (size_t)KVTOT + krow0 + (size_t)kc * BKV + kvc;
      bf16_t* lv = &Vs[kvr][kvc];
      async_g2l_b128(lv, gv);
      async_g2l_b128(lv + 8, gv + 8);
    }
    S_WAIT_ASYNC0();
    __syncthreads();  // tiles resident for all waves

    // S = Q * K^T : B operand lane n = kv column -> row n of Ks (contiguous d)
    v8f sacc[4];
#pragma unroll
    for (int nt = 0; nt < 4; ++nt) {
      const bf16_t* kp = &Ks[nt * 16 + lr16][0];
      v8f z = {};
      z = wmma_bf16(qa0, load_b_frag(kp, sel), z);
      z = wmma_bf16(qa1, load_b_frag(kp + 32, sel), z);
      sacc[nt] = z;
    }

    // Online softmax in registers. Row (i + 8*sel) spans lanes of one half;
    // xor masks 1..8 stay inside each 16-lane half (wave32-safe).
    float mnew[8], corr[8], psum[8];
#pragma unroll
    for (int i = 0; i < 8; ++i) {
      float mx = fmaxf(fmaxf(sacc[0][i], sacc[1][i]),
                       fmaxf(sacc[2][i], sacc[3][i]));
#pragma unroll
      for (int msk = 1; msk < 16; msk <<= 1)
        mx = fmaxf(mx, __shfl_xor(mx, msk, 32));
      mx *= scale;
      mnew[i] = fmaxf(m_run[i], mx);
      corr[i] = __expf(m_run[i] - mnew[i]);
      psum[i] = 0.0f;
    }
#pragma unroll
    for (int nt = 0; nt < 4; ++nt)
#pragma unroll
      for (int i = 0; i < 8; ++i) {
        const float p = __expf(sacc[nt][i] * scale - mnew[i]);
        psum[i] += p;
        Ps[wv][i + 8 * sel][nt * 16 + lr16] = f2bf(p);
      }
#pragma unroll
    for (int i = 0; i < 8; ++i) {
#pragma unroll
      for (int msk = 1; msk < 16; msk <<= 1)
        psum[i] += __shfl_xor(psum[i], msk, 32);
      l_run[i] = l_run[i] * corr[i] + psum[i];
      m_run[i] = mnew[i];
    }
#pragma unroll
    for (int nt = 0; nt < 4; ++nt)
#pragma unroll
      for (int i = 0; i < 8; ++i) ov[nt][i] *= corr[i];

    __syncthreads();  // P LDS visibility before A-fragment reads

    // O += P * V : A = P (16 x 64 kv), B lane n = d column -> row n of Vs
    const bf16_t* pp = &Ps[wv][lr16][0];
    v16bf pa0 = load_a_frag(pp, sel);       // kv 0..31
    v16bf pa1 = load_a_frag(pp + 32, sel);  // kv 32..63
#pragma unroll
    for (int nt = 0; nt < 4; ++nt) {
      const bf16_t* vp = &Vs[nt * 16 + lr16][0];
      ov[nt] = wmma_bf16(pa0, load_b_frag(vp, sel), ov[nt]);
      ov[nt] = wmma_bf16(pa1, load_b_frag(vp + 32, sel), ov[nt]);
    }
  }

  // Normalize and store bf16 attention output at [b, q, h, d]
#pragma unroll
  for (int i = 0; i < 8; ++i) l_run[i] = 1.0f / l_run[i];
#pragma unroll
  for (int nt = 0; nt < 4; ++nt)
#pragma unroll
    for (int i = 0; i < 8; ++i) {
      const int r = wv * 16 + i + 8 * sel;
      const int c = nt * 16 + lr16;
      Og[(qrow0 + r) * E + hcol + c] = f2bf(ov[nt][i] * l_run[i]);
    }
}

// ---------------------------------------------------------------------------
// Orchestration
// ---------------------------------------------------------------------------
extern "C" void kernel_launch(void* const* d_in, const int* in_sizes, int n_in,
                              void* d_out, int out_size, void* d_ws,
                              size_t ws_size, hipStream_t stream) {
  (void)in_sizes; (void)n_in; (void)out_size; (void)ws_size;
  constexpr int Bz = 2, SQ = 1024, SKV = 2048, E = 1024, VOC = 4096, H = 16;
  const int MQ = Bz * SQ;   // 2048 query rows
  const int MK = Bz * SKV;  // 4096 kv rows

  const float* x   = (const float*)d_in[0];
  const float* ctx = (const float*)d_in[1];
  const float* Wq  = (const float*)d_in[2];
  const float* bq  = (const float*)d_in[3];
  const float* Wk  = (const float*)d_in[4];
  const float* bk  = (const float*)d_in[5];
  const float* Wv  = (const float*)d_in[6];
  const float* bv  = (const float*)d_in[7];
  const float* Wp  = (const float*)d_in[8];
  const float* bp  = (const float*)d_in[9];
  float* out = (float*)d_out;

  char* ws = (char*)d_ws;
  size_t off = 0;
  auto carve = [&](size_t bytes) -> void* {
    void* p = ws + off;
    off += (bytes + 255) & ~(size_t)255;
    return p;
  };
  bf16_t* xbf = (bf16_t*)carve((size_t)MQ * E * sizeof(bf16_t));
  bf16_t* cbf = (bf16_t*)carve((size_t)MK * E * sizeof(bf16_t));
  bf16_t* wqT = (bf16_t*)carve((size_t)E * E * sizeof(bf16_t));
  bf16_t* wkT = (bf16_t*)carve((size_t)E * E * sizeof(bf16_t));
  bf16_t* wvT = (bf16_t*)carve((size_t)E * E * sizeof(bf16_t));
  bf16_t* wpT = (bf16_t*)carve((size_t)E * VOC * sizeof(bf16_t));
  bf16_t* Qb  = (bf16_t*)carve((size_t)MQ * E * sizeof(bf16_t));
  bf16_t* Kb  = (bf16_t*)carve((size_t)MK * E * sizeof(bf16_t));
  bf16_t* VtB = (bf16_t*)carve((size_t)E * MK * sizeof(bf16_t));
  bf16_t* AOb = (bf16_t*)carve((size_t)MQ * E * sizeof(bf16_t));

  cvt_f32_to_bf16<<<2048, 256, 0, stream>>>(x, xbf, (long long)MQ * E);
  cvt_f32_to_bf16<<<2048, 256, 0, stream>>>(ctx, cbf, (long long)MK * E);
  cvt_transpose_bf16<<<2048, 256, 0, stream>>>(Wq, wqT, E, E);
  cvt_transpose_bf16<<<2048, 256, 0, stream>>>(Wk, wkT, E, E);
  cvt_transpose_bf16<<<2048, 256, 0, stream>>>(Wv, wvT, E, E);
  cvt_transpose_bf16<<<4096, 256, 0, stream>>>(Wp, wpT, E, VOC);

  // Q = x*Wq+bq ; K = ctx*Wk+bk ; V = ctx*Wv+bv (stored transposed [E][MK])
  gemm_bf16_wmma<false, false><<<dim3(E / BN, MQ / BM), 256, 0, stream>>>(
      xbf, wqT, bq, Qb, MQ, E, E);
  gemm_bf16_wmma<false, false><<<dim3(E / BN, MK / BM), 256, 0, stream>>>(
      cbf, wkT, bk, Kb, MK, E, E);
  gemm_bf16_wmma<false, true><<<dim3(E / BN, MK / BM), 256, 0, stream>>>(
      cbf, wvT, bv, VtB, MK, E, E);

  // Fused softmax(Q K^T / sqrt(64)) V
  attn_fwd_wmma<<<dim3(SQ / BQ, H, Bz), 256, 0, stream>>>(
      Qb, Kb, VtB, AOb, SQ, SKV, E, MK, 0.125f);

  // out = AO * Wp + bp (f32 output)
  gemm_bf16_wmma<true, false><<<dim3(VOC / BN, MQ / BM), 256, 0, stream>>>(
      AOb, wpT, bp, out, MQ, VOC, E);
}
